// MessagePassingNetwork_52372831207656
// MI455X (gfx1250) — matrix operand order repaired and checked by reading, not compiled, not run on hardware
//
#include <hip/hip_runtime.h>
#include <hip/hip_bf16.h>

// ---------------------------------------------------------------------------
// MessagePassingNetwork for MI455X (gfx1250, wave32, WMMA)
//  - weights pre-transposed to bf16 [N][K]
//  - per-block weight slices staged to LDS via global_load_async_to_lds_b128
//    (ASYNCcnt) once per K-panel; WMMA fed by ds_load_b128 fragments
//  - dense GEMMs use v_wmma_f32_16x16x32_bf16, 2 N-tiles per wave
//  - edge phase is a memory-bound scalar kernel with hw f32 atomics
// ---------------------------------------------------------------------------

typedef __attribute__((ext_vector_type(16))) __bf16 v16bf;
typedef __attribute__((ext_vector_type(8)))  float  v8f;

#define EMB_D 256
#define MSG_D 64
#define MAX_L 8

__device__ __forceinline__ float silu_f(float x) {
    return x * __builtin_amdgcn_rcpf(1.0f + __expf(-x));
}

__device__ __forceinline__ v8f wmma_bf16(v16bf a, v16bf b, v8f c) {
    return __builtin_amdgcn_wmma_f32_16x16x32_bf16(
        false, a, false, b, (short)0, c, false, false);
}

// Issue one async 16B global->LDS copy (ASYNCcnt-tracked).
__device__ __forceinline__ void async_copy16(uint32_t ldsAddr, const void* gaddr) {
    asm volatile("global_load_async_to_lds_b128 %0, %1, off"
                 :: "v"(ldsAddr), "v"(gaddr) : "memory");
}

__device__ __forceinline__ void wait_async0() {
    asm volatile("s_wait_asynccnt 0x0" ::: "memory");
}

// Cooperatively stage 64 rows x KP bf16 cols (row stride ldg) into LDS.
// total 16B-chunks is a multiple of 256 for all shapes used -> no divergence.
__device__ __forceinline__ void stage_rows_async(const __bf16* __restrict__ g,
                                                 int ldg, uint32_t ldsBase,
                                                 int KP, int tid) {
    const int chunksPerRow = KP >> 3;            // 16B = 8 bf16
    const int total = 64 * chunksPerRow;
    for (int idx = tid; idx < total; idx += 256) {
        const int row = idx / chunksPerRow;
        const int ch  = idx - row * chunksPerRow;
        const void*   ga = g + (size_t)row * ldg + (ch << 3);
        const uint32_t la = ldsBase + (uint32_t)(((row * KP) + (ch << 3)) << 1);
        async_copy16(la, ga);
    }
}

// Build the A fragment (16 contiguous k per lane half: k = 16h + s).
__device__ __forceinline__ v16bf load_a_frag(const float* ap, float scale) {
    const float4 a0 = *(const float4*)(ap);
    const float4 a1 = *(const float4*)(ap + 4);
    const float4 a2 = *(const float4*)(ap + 8);
    const float4 a3 = *(const float4*)(ap + 12);
    v16bf af;
    af[0]  = (__bf16)(a0.x * scale); af[1]  = (__bf16)(a0.y * scale);
    af[2]  = (__bf16)(a0.z * scale); af[3]  = (__bf16)(a0.w * scale);
    af[4]  = (__bf16)(a1.x * scale); af[5]  = (__bf16)(a1.y * scale);
    af[6]  = (__bf16)(a1.z * scale); af[7]  = (__bf16)(a1.w * scale);
    af[8]  = (__bf16)(a2.x * scale); af[9]  = (__bf16)(a2.y * scale);
    af[10] = (__bf16)(a2.z * scale); af[11] = (__bf16)(a2.w * scale);
    af[12] = (__bf16)(a3.x * scale); af[13] = (__bf16)(a3.y * scale);
    af[14] = (__bf16)(a3.z * scale); af[15] = (__bf16)(a3.w * scale);
    return af;
}

// ---------------------------------------------------------------------------
// GEMM: C[M,N] = act( (rowScale?rowScale[m]:1) * A[M,K] @ B[K,N] + bias[N] )
// Bt = bf16 transposed weights [N][K].  Block = 8 waves (4 mRows x 2 nCols),
// each wave computes a 16x32 strip (2 accumulators). Block tile 64x64.
// B slice (64 cols x KP) staged in LDS per K-panel, KP = min(K,256).
// Dynamic LDS: 64*KP*2 bytes.  Grid: x=ceil(M/64), y=N/64.
// ---------------------------------------------------------------------------
template <int ACT>
__global__ __launch_bounds__(256)
void gemm_wmma_kernel(const float*  __restrict__ A, int lda,
                      const __bf16* __restrict__ Bt,
                      const float*  __restrict__ bias,
                      const float*  __restrict__ rowScale,
                      float* __restrict__ C, int ldc,
                      int M, int N, int K)
{
    extern __shared__ char smem_raw[];
    __bf16* sB = (__bf16*)smem_raw;
    const uint32_t sBase = (uint32_t)(uintptr_t)smem_raw;   // LDS byte address

    const int tid     = threadIdx.x;
    const int lane    = tid & 31;
    const int wave    = tid >> 5;            // 0..7
    const int waveRow = wave >> 1;           // 0..3
    const int waveCol = wave & 1;            // 0..1
    const int mBase   = blockIdx.x * 64 + waveRow * 16;
    const int nBlock  = blockIdx.y * 64;

    const int half = lane >> 4;              // 0 or 1
    const int sub  = lane & 15;

    int rowA = mBase + sub;
    if (rowA >= M) rowA = M - 1;             // clamp loads; stores predicated
    const float scale = rowScale ? rowScale[rowA] : 1.0f;
    const float* aRow = A + (size_t)rowA * lda;

    const int KP  = (K < 256) ? K : 256;     // K-panel staged in LDS
    const int lc0 = waveCol * 32 + sub;      // local B row (= output col) 0
    const int lc1 = lc0 + 16;

    v8f acc0 = {}, acc1 = {};
    for (int kp = 0; kp < K; kp += KP) {
        stage_rows_async(Bt + (size_t)nBlock * K + kp, K, sBase, KP, tid);
        wait_async0();
        __syncthreads();

        const __bf16* s0 = sB + lc0 * KP + half * 16;
        const __bf16* s1 = sB + lc1 * KP + half * 16;
        for (int kk = 0; kk < KP; kk += 32) {
            const v16bf af  = load_a_frag(aRow + kp + kk + half * 16, scale);
            const v16bf bf0 = *(const v16bf*)(s0 + kk);     // 2x ds_load_b128
            const v16bf bf1 = *(const v16bf*)(s1 + kk);
            acc0 = wmma_bf16(af, bf0, acc0);
            acc1 = wmma_bf16(af, bf1, acc1);
        }
        __syncthreads();                     // before next panel overwrites sB
    }

    const int   col0 = nBlock + lc0;
    const int   col1 = nBlock + lc1;
    const float bb0  = bias ? bias[col0] : 0.0f;
    const float bb1  = bias ? bias[col1] : 0.0f;
    #pragma unroll
    for (int r = 0; r < 8; ++r) {
        const int row = mBase + r + half * 8;   // C layout: VGPR r -> M=r / 8+r
        if (row < M) {
            float v0 = acc0[r] + bb0;
            float v1 = acc1[r] + bb1;
            if (ACT == 1) { v0 = silu_f(v0); v1 = silu_f(v1); }
            C[(size_t)row * ldc + col0] = v0;
            C[(size_t)row * ldc + col1] = v1;
        }
    }
}

// ---------------------------------------------------------------------------
// Fused GLU: C = (A@Wv) * silu(A@Wg + bg) + addin        (K = 256 here)
// Both weight slices staged in LDS, K-panels of 128 -> 32KB dynamic LDS.
// A fragment shared across 4 WMMAs (2 weights x 2 N-tiles).
// ---------------------------------------------------------------------------
__global__ __launch_bounds__(256)
void glu_wmma_kernel(const float*  __restrict__ A, int lda,
                     const __bf16* __restrict__ WvT,
                     const __bf16* __restrict__ WgT,
                     const float*  __restrict__ bg,
                     const float*  __restrict__ addin, int ldadd,
                     float* __restrict__ C, int ldc,
                     int M, int N, int K)
{
    constexpr int KP = 128;
    extern __shared__ char smem_raw[];
    __bf16* sV = (__bf16*)smem_raw;                      // [64][KP]
    __bf16* sG = sV + 64 * KP;                           // [64][KP]
    const uint32_t sBase = (uint32_t)(uintptr_t)smem_raw;

    const int tid     = threadIdx.x;
    const int lane    = tid & 31;
    const int wave    = tid >> 5;
    const int waveRow = wave >> 1;
    const int waveCol = wave & 1;
    const int mBase   = blockIdx.x * 64 + waveRow * 16;
    const int nBlock  = blockIdx.y * 64;

    const int half = lane >> 4;
    const int sub  = lane & 15;

    int rowA = mBase + sub;
    if (rowA >= M) rowA = M - 1;
    const float* aRow = A + (size_t)rowA * lda;

    const int lc0 = waveCol * 32 + sub;
    const int lc1 = lc0 + 16;

    v8f accV0 = {}, accV1 = {}, accG0 = {}, accG1 = {};
    for (int kp = 0; kp < K; kp += KP) {
        stage_rows_async(WvT + (size_t)nBlock * K + kp, K, sBase, KP, tid);
        stage_rows_async(WgT + (size_t)nBlock * K + kp, K,
                         sBase + 64 * KP * 2, KP, tid);
        wait_async0();
        __syncthreads();

        const __bf16* v0p = sV + lc0 * KP + half * 16;
        const __bf16* v1p = sV + lc1 * KP + half * 16;
        const __bf16* g0p = sG + lc0 * KP + half * 16;
        const __bf16* g1p = sG + lc1 * KP + half * 16;
        for (int kk = 0; kk < KP; kk += 32) {
            const v16bf af = load_a_frag(aRow + kp + kk + half * 16, 1.0f);
            const v16bf v0 = *(const v16bf*)(v0p + kk);
            const v16bf v1 = *(const v16bf*)(v1p + kk);
            const v16bf g0 = *(const v16bf*)(g0p + kk);
            const v16bf g1 = *(const v16bf*)(g1p + kk);
            accV0 = wmma_bf16(af, v0, accV0);
            accV1 = wmma_bf16(af, v1, accV1);
            accG0 = wmma_bf16(af, g0, accG0);
            accG1 = wmma_bf16(af, g1, accG1);
        }
        __syncthreads();
    }

    const int   col0 = nBlock + lc0;
    const int   col1 = nBlock + lc1;
    const float bb0  = bg[col0];
    const float bb1  = bg[col1];
    #pragma unroll
    for (int r = 0; r < 8; ++r) {
        const int row = mBase + r + half * 8;
        if (row < M) {
            const float* arow = addin + (size_t)row * ldadd;
            C[(size_t)row * ldc + col0] =
                accV0[r] * silu_f(accG0[r] + bb0) + arow[col0];
            C[(size_t)row * ldc + col1] =
                accV1[r] * silu_f(accG1[r] + bb1) + arow[col1];
        }
    }
}

// ---------------------------------------------------------------------------
// Edge phase: h = s_proj[snd] + r_proj[rcv]; layernorm; silu; * gate;
// atomic segment-sum into msg_accum[rcv].  Memory bound -> thread per edge.
// ---------------------------------------------------------------------------
__global__ __launch_bounds__(256)
void edge_kernel(const float* __restrict__ s_proj,
                 const float* __restrict__ r_proj,
                 const int*   __restrict__ senders,
                 const int*   __restrict__ receivers,
                 const float* __restrict__ gate,       // [edgeCount, 64]
                 const float* __restrict__ ln_scale,   // [64]
                 const float* __restrict__ ln_bias,    // [64]
                 float* __restrict__ msg_accum,        // [n_nodes, 64]
                 int edgeBase, int edgeCount)
{
    const int i = blockIdx.x * blockDim.x + threadIdx.x;
    if (i >= edgeCount) return;
    const int e = edgeBase + i;
    const int s = senders[e];
    const int r = receivers[e];

    const float4* sp = (const float4*)(s_proj + (size_t)s * MSG_D);
    const float4* rp = (const float4*)(r_proj + (size_t)r * MSG_D);

    float h[MSG_D];
    float sum = 0.0f;
    #pragma unroll
    for (int j = 0; j < MSG_D / 4; ++j) {
        float4 a = sp[j];
        float4 b = rp[j];
        float x0 = a.x + b.x, x1 = a.y + b.y, x2 = a.z + b.z, x3 = a.w + b.w;
        h[4*j+0] = x0; h[4*j+1] = x1; h[4*j+2] = x2; h[4*j+3] = x3;
        sum += x0 + x1 + x2 + x3;
    }
    const float mean = sum * (1.0f / MSG_D);
    float var = 0.0f;
    #pragma unroll
    for (int j = 0; j < MSG_D; ++j) {
        const float d = h[j] - mean;
        var += d * d;
    }
    var *= (1.0f / MSG_D);
    const float rstd = rsqrtf(var + 1e-6f);

    const float* gp  = gate + (size_t)i * MSG_D;
    float*       dst = msg_accum + (size_t)r * MSG_D;
    #pragma unroll
    for (int j = 0; j < MSG_D; ++j) {
        float t = (h[j] - mean) * rstd * ln_scale[j] + ln_bias[j];
        t = silu_f(t) * gp[j];
        __hip_atomic_fetch_add(dst + j, t, __ATOMIC_RELAXED,
                               __HIP_MEMORY_SCOPE_AGENT);   // global_atomic_add_f32
    }
}

// ---------------------------------------------------------------------------
// Utility kernels
// ---------------------------------------------------------------------------
__global__ void fill_kernel(float* __restrict__ p, float v, int n) {
    const int i = blockIdx.x * blockDim.x + threadIdx.x;
    if (i < n) p[i] = v;
}

__global__ void norm_accum_kernel(float* __restrict__ acc,
                                  const float* __restrict__ e_norm,
                                  const int* __restrict__ receivers, int n) {
    const int i = blockIdx.x * blockDim.x + threadIdx.x;
    if (i < n)
        __hip_atomic_fetch_add(acc + receivers[i], e_norm[i],
                               __ATOMIC_RELAXED, __HIP_MEMORY_SCOPE_AGENT);
}

__global__ void recip_kernel(float* __restrict__ p, int n) {
    const int i = blockIdx.x * blockDim.x + threadIdx.x;
    if (i < n) p[i] = 1.0f / p[i];
}

// W [K,N] f32  ->  Wt [N,K] bf16
__global__ void transpose_bf16_kernel(const float* __restrict__ W,
                                      __bf16* __restrict__ Wt,
                                      int K, int N) {
    const int idx = blockIdx.x * blockDim.x + threadIdx.x;
    if (idx >= K * N) return;
    const int n = idx / K;
    const int k = idx - n * K;
    Wt[(size_t)n * K + k] = (__bf16)W[(size_t)k * N + n];
}

// ---------------------------------------------------------------------------
// Host orchestration
// ---------------------------------------------------------------------------
extern "C" void kernel_launch(void* const* d_in, const int* in_sizes, int n_in,
                              void* d_out, int out_size, void* d_ws, size_t ws_size,
                              hipStream_t stream)
{
    const float* s_embed  = (const float*)d_in[0];
    const float* r_embed0 = (const float*)d_in[1];
    const float* e_embed  = (const float*)d_in[2];
    const float* e_norm   = (const float*)d_in[3];
    const int*   senders   = (const int*)d_in[4];
    const int*   receivers = (const int*)d_in[5];
    const float* Ws    = (const float*)d_in[6];
    const float* bs    = (const float*)d_in[7];
    const float* Wr    = (const float*)d_in[8];
    const float* br    = (const float*)d_in[9];
    const float* lns   = (const float*)d_in[10];
    const float* lnb   = (const float*)d_in[11];
    const float* We    = (const float*)d_in[12];
    const float* Wmsg  = (const float*)d_in[13];
    const float* g1_Wv = (const float*)d_in[14];
    const float* g1_Wg = (const float*)d_in[15];
    const float* g1_bg = (const float*)d_in[16];
    const float* g2_Wv = (const float*)d_in[17];
    const float* g2_Wg = (const float*)d_in[18];
    const float* g2_bg = (const float*)d_in[19];
    const float* Wf    = (const float*)d_in[20];
    const float* bfin  = (const float*)d_in[21];
    float* out = (float*)d_out;

    const int n_nodes = in_sizes[0] / EMB_D;
    const int n_edges = in_sizes[3];
    int L = in_sizes[6] / (EMB_D * MSG_D);
    if (L > MAX_L) L = MAX_L;

    // ---- workspace carve-up ----
    float* ws = (float*)d_ws;
    size_t off = 0;
    auto carve = [&](size_t n) { float* p = ws + off; off += n; return p; };
    float* norm      = carve((size_t)n_nodes);
    float* s_proj    = carve((size_t)n_nodes * MSG_D);
    float* r_proj    = carve((size_t)n_nodes * MSG_D);
    float* msg_accum = carve((size_t)n_nodes * MSG_D);
    float* msg_act   = carve((size_t)n_nodes * EMB_D);
    float* ybuf      = carve((size_t)n_nodes * EMB_D);
    float* embcat    = carve((size_t)n_nodes * EMB_D * L);
    const int CHUNK  = 131072;
    float* gatebuf   = carve((size_t)CHUNK * MSG_D);

    // bf16 weight region (32B aligned: pad to 8 floats)
    off = (off + 7) & ~(size_t)7;
    __bf16* wb = (__bf16*)(ws + off);
    size_t boff = 0;
    auto carveB = [&](size_t n) { __bf16* p = wb + boff; boff += n; return p; };
    __bf16 *WsT[MAX_L], *WrT[MAX_L], *WeT[MAX_L], *WmsgT[MAX_L];
    __bf16 *g1WvT[MAX_L], *g1WgT[MAX_L], *g2WvT[MAX_L], *g2WgT[MAX_L];
    for (int l = 0; l < L; ++l) {
        WsT[l]   = carveB((size_t)EMB_D * MSG_D);
        WrT[l]   = carveB((size_t)EMB_D * MSG_D);
        WeT[l]   = carveB((size_t)MSG_D * MSG_D);
        WmsgT[l] = carveB((size_t)MSG_D * EMB_D);
        g1WvT[l] = carveB((size_t)EMB_D * EMB_D);
        g1WgT[l] = carveB((size_t)EMB_D * EMB_D);
        g2WvT[l] = carveB((size_t)EMB_D * EMB_D);
        g2WgT[l] = carveB((size_t)EMB_D * EMB_D);
    }
    __bf16* WfT = carveB((size_t)L * EMB_D * EMB_D);
    (void)ws_size; (void)n_in; (void)out_size;

    const int  ldcat = EMB_D * L;
    const dim3 blk(256);

    // ---- pre-transpose all weights to bf16 [N][K] ----
    auto tpose = [&](const float* W, __bf16* Wt, int K, int N) {
        const int tot = K * N;
        transpose_bf16_kernel<<<(tot + 255) / 256, blk, 0, stream>>>(W, Wt, K, N);
    };
    for (int l = 0; l < L; ++l) {
        tpose(Ws   + (size_t)l * EMB_D * MSG_D, WsT[l],   EMB_D, MSG_D);
        tpose(Wr   + (size_t)l * EMB_D * MSG_D, WrT[l],   EMB_D, MSG_D);
        tpose(We   + (size_t)l * MSG_D * MSG_D, WeT[l],   MSG_D, MSG_D);
        tpose(Wmsg + (size_t)l * MSG_D * EMB_D, WmsgT[l], MSG_D, EMB_D);
        tpose(g1_Wv + (size_t)l * EMB_D * EMB_D, g1WvT[l], EMB_D, EMB_D);
        tpose(g1_Wg + (size_t)l * EMB_D * EMB_D, g1WgT[l], EMB_D, EMB_D);
        tpose(g2_Wv + (size_t)l * EMB_D * EMB_D, g2WvT[l], EMB_D, EMB_D);
        tpose(g2_Wg + (size_t)l * EMB_D * EMB_D, g2WgT[l], EMB_D, EMB_D);
    }
    tpose(Wf, WfT, ldcat, EMB_D);

    // ---- per-receiver normalizer: 1 / (segment_sum(e_norm) + 1) ----
    fill_kernel<<<(n_nodes + 255) / 256, blk, 0, stream>>>(norm, 1.0f, n_nodes);
    norm_accum_kernel<<<(n_edges + 255) / 256, blk, 0, stream>>>(
        norm, e_norm, receivers, n_edges);
    recip_kernel<<<(n_nodes + 255) / 256, blk, 0, stream>>>(norm, n_nodes);

    const dim3 gN64((n_nodes + 63) / 64, MSG_D / 64);   // N = 64
    const dim3 gN256((n_nodes + 63) / 64, EMB_D / 64);  // N = 256
    auto smemG = [](int K) { return (size_t)64 * ((K < 256) ? K : 256) * 2; };
    const size_t smemGLU = (size_t)2 * 64 * 128 * 2;     // 32KB

    for (int l = 0; l < L; ++l) {
        const float* r_cur;
        int ldr;
        if (l == 0) { r_cur = r_embed0;                 ldr = EMB_D; }
        else        { r_cur = embcat + (l - 1) * EMB_D; ldr = ldcat; }
        float* r_out = embcat + l * EMB_D;              // ldc = ldcat

        // s_proj = s_embed @ Ws[l] + bs[l]
        gemm_wmma_kernel<0><<<gN64, blk, smemG(EMB_D), stream>>>(
            s_embed, EMB_D, WsT[l], bs + (size_t)l * MSG_D, nullptr,
            s_proj, MSG_D, n_nodes, MSG_D, EMB_D);
        // r_proj = r_cur @ Wr[l] + br[l]
        gemm_wmma_kernel<0><<<gN64, blk, smemG(EMB_D), stream>>>(
            r_cur, ldr, WrT[l], br + (size_t)l * MSG_D, nullptr,
            r_proj, MSG_D, n_nodes, MSG_D, EMB_D);

        // msg_accum = 0
        fill_kernel<<<((size_t)n_nodes * MSG_D + 255) / 256, blk, 0, stream>>>(
            msg_accum, 0.0f, n_nodes * MSG_D);

        // edge phase, chunked: gate GEMM (WMMA) then gather/LN/silu/scatter
        for (int base = 0; base < n_edges; base += CHUNK) {
            const int cnt = (n_edges - base < CHUNK) ? (n_edges - base) : CHUNK;
            gemm_wmma_kernel<0><<<dim3((cnt + 63) / 64, 1), blk,
                                  smemG(MSG_D), stream>>>(
                e_embed + (size_t)base * MSG_D, MSG_D, WeT[l],
                nullptr, nullptr, gatebuf, MSG_D, cnt, MSG_D, MSG_D);
            edge_kernel<<<(cnt + 255) / 256, blk, 0, stream>>>(
                s_proj, r_proj, senders, receivers, gatebuf,
                lns + (size_t)l * MSG_D, lnb + (size_t)l * MSG_D,
                msg_accum, base, cnt);
        }

        // msg_act = silu((msg_accum * norm[:,None]) @ Wmsg[l])
        gemm_wmma_kernel<1><<<gN256, blk, smemG(MSG_D), stream>>>(
            msg_accum, MSG_D, WmsgT[l], nullptr, norm,
            msg_act, EMB_D, n_nodes, EMB_D, MSG_D);

        // y = (r_cur @ g1_Wv) * silu(r_cur @ g1_Wg + g1_bg) + msg_act
        glu_wmma_kernel<<<gN256, blk, smemGLU, stream>>>(
            r_cur, ldr, g1WvT[l], g1WgT[l], g1_bg + (size_t)l * EMB_D,
            msg_act, EMB_D, ybuf, EMB_D, n_nodes, EMB_D, EMB_D);

        // r_out = (y @ g2_Wv) * silu(y @ g2_Wg + g2_bg) + r_cur
        glu_wmma_kernel<<<gN256, blk, smemGLU, stream>>>(
            ybuf, EMB_D, g2WvT[l], g2WgT[l], g2_bg + (size_t)l * EMB_D,
            r_cur, ldr, r_out, ldcat, n_nodes, EMB_D, EMB_D);
    }

    // out = embcat @ Wf + bf
    gemm_wmma_kernel<0><<<gN256, blk, smemG(ldcat), stream>>>(
        embcat, ldcat, WfT, bfin, nullptr,
        out, EMB_D, n_nodes, EMB_D, ldcat);
}
